// VideoAttn_70506183131463
// MI455X (gfx1250) — matrix operand order repaired and checked by reading, not compiled, not run on hardware
//
#include <hip/hip_runtime.h>
#include <hip/hip_bf16.h>

// ---------------------------------------------------------------------------
// VideoAttn on MI455X (gfx1250): f16 WMMA (f32 accum) pipeline.
//   x:(4800,1024) -> Q/K/V gemm (64x64 tiles) -> RMSNorm+RoPE ->
//   flash attention (32 queries/wave) -> out proj
// ---------------------------------------------------------------------------

#define L_TOK  4800
#define DMODEL 1024
#define NHEADS 16
#define DHEAD  64

typedef __attribute__((ext_vector_type(16))) _Float16 v16h;
typedef __attribute__((ext_vector_type(8)))  _Float16 v8h;
typedef __attribute__((ext_vector_type(8)))  float    v8f;

#define WMMA(a, b, c) \
  __builtin_amdgcn_wmma_f32_16x16x32_f16(false, (a), false, (b), (short)0, (c), false, false)

// -------------------------- fragment loaders -------------------------------
// A (16x32 f16, contiguous k order, same as B): lane m=lane&15 holds row m.
//   halves j=0..15 -> K=kb+j, kb=k0+(lane>=16?16:0).
__device__ __forceinline__ v16h load_a_frag(const _Float16* A, int lda, int row0,
                                            int k0, int lane) {
  int m  = lane & 15;
  int kb = k0 + ((lane >> 4) << 4);
  const _Float16* p = A + (size_t)(row0 + m) * lda + kb;
  union { v16h v; v8h h[2]; } u;
  u.h[0] = *(const v8h*)(p);
  u.h[1] = *(const v8h*)(p + 8);
  return u.v;
}

// B (32x16 f16): element(K,N); lane n=lane&15 -> column N, halves j=0..15 ->
//   K = k0 + (lane>=16?16:0) + j.  Source stored N-major => 32B/lane load.
__device__ __forceinline__ v16h load_b_frag(const _Float16* Bt, int ldb, int n0,
                                            int k0, int lane) {
  int n  = lane & 15;
  int kb = k0 + ((lane >> 4) << 4);
  return *(const v16h*)(Bt + (size_t)(n0 + n) * ldb + kb);
}

// ----------------------------- kernels -------------------------------------
__global__ __launch_bounds__(256) void f32_to_f16_kernel(const float* __restrict__ in,
                                                         _Float16* __restrict__ out,
                                                         int n) {
  int i = blockIdx.x * 256 + threadIdx.x;
  if (i < n) out[i] = (_Float16)in[i];
}

// C[r,c] = sum_k A[r,k] * Bt[c,k];  A:(rows,K) f16, Bt:(cols,K) f16, C f32.
// One wave -> 64x64 tile: 8 fragment loads : 16 WMMAs per 32-wide K step.
__global__ __launch_bounds__(256) void gemm_f16_nt(const _Float16* __restrict__ A,
                                                   const _Float16* __restrict__ Bt,
                                                   float* __restrict__ C,
                                                   int rows, int cols, int K) {
  int lane = threadIdx.x & 31;
  int wid  = threadIdx.x >> 5;
  int colTiles = cols >> 6;
  int tiles    = (rows >> 6) * colTiles;
  int tile = blockIdx.x * (blockDim.x >> 5) + wid;
  if (tile >= tiles) return;                 // wave-uniform: EXEC stays all-1s
  int r0 = (tile / colTiles) << 6;
  int c0 = (tile % colTiles) << 6;

  v8f zero = {};
  v8f acc[4][4];
#pragma unroll
  for (int i = 0; i < 4; i++)
#pragma unroll
    for (int j = 0; j < 4; j++) acc[i][j] = zero;

#pragma unroll 1
  for (int k0 = 0; k0 < K; k0 += 32) {
    v16h a[4], b[4];
#pragma unroll
    for (int i = 0; i < 4; i++) a[i] = load_a_frag(A, K, r0 + 16 * i, k0, lane);
#pragma unroll
    for (int j = 0; j < 4; j++) b[j] = load_b_frag(Bt, K, c0 + 16 * j, k0, lane);
#pragma unroll
    for (int i = 0; i < 4; i++)
#pragma unroll
      for (int j = 0; j < 4; j++) acc[i][j] = WMMA(a[i], b[j], acc[i][j]);
  }

  int n = lane & 15, mb = (lane >> 4) << 3;   // C layout: lanes0-15 rows 0-7
#pragma unroll
  for (int i = 0; i < 4; i++)
#pragma unroll
    for (int j = 0; j < 4; j++)
#pragma unroll
      for (int v = 0; v < 8; v++)
        C[(size_t)(r0 + 16 * i + mb + v) * cols + c0 + 16 * j + n] = acc[i][j][v];
}

// Per-head RMSNorm (eps 1e-6) + RoPE(rotate-half) + scale; one wave per (l,h).
// dst layout: (head, L, Dh) f16.
__global__ __launch_bounds__(256) void qk_post_kernel(const float* __restrict__ S,
                                                      const float* __restrict__ rope,
                                                      const float* __restrict__ gamma,
                                                      _Float16* __restrict__ dst,
                                                      float scale) {
  int lane = threadIdx.x & 31;
  int wid  = threadIdx.x >> 5;
  int idx  = blockIdx.x * (blockDim.x >> 5) + wid;
  if (idx >= L_TOK * NHEADS) return;
  int l = idx >> 4, h = idx & 15;

  const float* src = S + (size_t)l * DMODEL + h * DHEAD;
  float t0 = src[lane], t1 = src[lane + 32];
  float ss = t0 * t0 + t1 * t1;
  ss += __shfl_xor(ss, 1, 32);
  ss += __shfl_xor(ss, 2, 32);
  ss += __shfl_xor(ss, 4, 32);
  ss += __shfl_xor(ss, 8, 32);
  ss += __shfl_xor(ss, 16, 32);
  float r  = rsqrtf(ss * (1.0f / DHEAD) + 1e-6f);
  float n0 = t0 * r * gamma[lane];
  float n1 = t1 * r * gamma[lane + 32];

  const float* re = rope + (size_t)l * DHEAD;
  float a0 = re[lane], a1 = re[lane + 32];
  // dh<32: out = n*cos - n[dh+32]*sin ; dh>=32: out = n*cos + n[dh-32]*sin
  float o0 = (n0 * __cosf(a0) - n1 * __sinf(a0)) * scale;
  float o1 = (n1 * __cosf(a1) + n0 * __sinf(a1)) * scale;

  _Float16* d = dst + ((size_t)h * L_TOK + l) * DHEAD;
  d[lane]      = (_Float16)o0;
  d[lane + 32] = (_Float16)o1;
}

// V (L,D) f32 -> (head, dh, L) f16 transpose via LDS 32x32 tiles (coalesced
// on both sides):  vt[c*L + l] = S[l*D + c].
__global__ __launch_bounds__(256) void v_trans_kernel(const float* __restrict__ S,
                                                      _Float16* __restrict__ vt) {
  __shared__ float tile[32][33];
  int tx = threadIdx.x & 31;
  int ty = threadIdx.x >> 5;           // 0..7
  int l0 = blockIdx.x * 32;
  int c0 = blockIdx.y * 32;
#pragma unroll
  for (int i = 0; i < 32; i += 8)
    tile[ty + i][tx] = S[(size_t)(l0 + ty + i) * DMODEL + c0 + tx];
  __syncthreads();
#pragma unroll
  for (int i = 0; i < 32; i += 8)
    vt[(size_t)(c0 + ty + i) * L_TOK + l0 + tx] = (_Float16)tile[tx][ty + i];
}

// Flash attention: one wave per (head, 32-query block). Q frags in registers,
// online softmax in f32, P re-shaped C->A layout through a 2KB LDS slice.
// 16 WMMAs per 8 global fragment loads per 32-key chunk.
__global__ __launch_bounds__(128) void flash_kernel(const _Float16* __restrict__ qh,
                                                    const _Float16* __restrict__ kh,
                                                    const _Float16* __restrict__ vt,
                                                    _Float16* __restrict__ ocat) {
  __shared__ __align__(32) _Float16 pbuf[4][32 * 32];
  const int lane = threadIdx.x & 31;
  const int wid  = threadIdx.x >> 5;
  const int idx  = blockIdx.x * 4 + wid;        // 0 .. 150*16-1
  const int h    = idx / (L_TOK / 32);
  const int q0   = (idx - h * (L_TOK / 32)) * 32;

  const _Float16* Q = qh + (size_t)h * L_TOK * DHEAD;   // (L,64), scale folded
  const _Float16* K = kh + (size_t)h * L_TOK * DHEAD;   // (L,64)
  const _Float16* V = vt + (size_t)h * DHEAD * L_TOK;   // (64,L)
  _Float16* P = pbuf[wid];
  const int n  = lane & 15;
  const int mb = (lane >> 4) << 3;

  v16h aq[2][2];
#pragma unroll
  for (int qt = 0; qt < 2; qt++) {
    aq[qt][0] = load_a_frag(Q, DHEAD, q0 + 16 * qt, 0, lane);
    aq[qt][1] = load_a_frag(Q, DHEAD, q0 + 16 * qt, 32, lane);
  }

  v8f zero = {};
  v8f o[2][4];
  float mrun[2][8], lrun[2][8];
#pragma unroll
  for (int qt = 0; qt < 2; qt++) {
#pragma unroll
    for (int t = 0; t < 4; t++) o[qt][t] = zero;
#pragma unroll
    for (int v = 0; v < 8; v++) { mrun[qt][v] = -1e30f; lrun[qt][v] = 0.0f; }
  }

#pragma unroll 1
  for (int k0 = 0; k0 < L_TOK; k0 += 32) {
    // S tiles: 32 queries x 32 keys, Dh reduced in two WMMAs per 16x16 tile.
    v16h b00 = load_b_frag(K, DHEAD, k0,      0,  lane);
    v16h b01 = load_b_frag(K, DHEAD, k0,      32, lane);
    v16h b10 = load_b_frag(K, DHEAD, k0 + 16, 0,  lane);
    v16h b11 = load_b_frag(K, DHEAD, k0 + 16, 32, lane);
    v8f s[2][2];
#pragma unroll
    for (int qt = 0; qt < 2; qt++) {
      v8f s0 = zero, s1 = zero;
      s0 = WMMA(aq[qt][0], b00, s0);
      s0 = WMMA(aq[qt][1], b01, s0);
      s1 = WMMA(aq[qt][0], b10, s1);
      s1 = WMMA(aq[qt][1], b11, s1);
      s[qt][0] = s0;
      s[qt][1] = s1;
    }

    // Online softmax per row (row lives in one VGPR across a 16-lane half).
#pragma unroll
    for (int qt = 0; qt < 2; qt++) {
#pragma unroll
      for (int v = 0; v < 8; v++) {
        float r = fmaxf(s[qt][0][v], s[qt][1][v]);
        r = fmaxf(r, __shfl_xor(r, 1, 32));
        r = fmaxf(r, __shfl_xor(r, 2, 32));
        r = fmaxf(r, __shfl_xor(r, 4, 32));
        r = fmaxf(r, __shfl_xor(r, 8, 32));
        float nm = fmaxf(mrun[qt][v], r);
        float al = __expf(mrun[qt][v] - nm);
        mrun[qt][v] = nm;
        float p0 = __expf(s[qt][0][v] - nm);
        float p1 = __expf(s[qt][1][v] - nm);
        float rs = p0 + p1;
        rs += __shfl_xor(rs, 1, 32);
        rs += __shfl_xor(rs, 2, 32);
        rs += __shfl_xor(rs, 4, 32);
        rs += __shfl_xor(rs, 8, 32);
        lrun[qt][v] = lrun[qt][v] * al + rs;
#pragma unroll
        for (int t = 0; t < 4; t++) o[qt][t][v] *= al;
        P[(16 * qt + mb + v) * 32 + n]      = (_Float16)p0;
        P[(16 * qt + mb + v) * 32 + 16 + n] = (_Float16)p1;
      }
    }
    __asm__ volatile("s_wait_dscnt 0x0" ::: "memory");  // P visible across lanes

    v16h bv0 = load_b_frag(V, L_TOK, 0,  k0, lane);     // V as B (32 keys x 16 dh)
    v16h bv1 = load_b_frag(V, L_TOK, 16, k0, lane);
    v16h bv2 = load_b_frag(V, L_TOK, 32, k0, lane);
    v16h bv3 = load_b_frag(V, L_TOK, 48, k0, lane);
#pragma unroll
    for (int qt = 0; qt < 2; qt++) {
      v16h pa = load_a_frag(P, 32, 16 * qt, 0, lane);   // P as A (16x32)
      o[qt][0] = WMMA(pa, bv0, o[qt][0]);
      o[qt][1] = WMMA(pa, bv1, o[qt][1]);
      o[qt][2] = WMMA(pa, bv2, o[qt][2]);
      o[qt][3] = WMMA(pa, bv3, o[qt][3]);
    }
    __asm__ volatile("" ::: "memory");  // keep next-iter P stores after the read
  }

  // Normalize and scatter into concat-head layout (L, D) f16 for the out-proj.
#pragma unroll
  for (int qt = 0; qt < 2; qt++) {
#pragma unroll
    for (int v = 0; v < 8; v++) {
      float inv = 1.0f / lrun[qt][v];
      size_t rowoff = (size_t)(q0 + 16 * qt + mb + v) * DMODEL + h * DHEAD + n;
#pragma unroll
      for (int t = 0; t < 4; t++)
        ocat[rowoff + t * 16] = (_Float16)(o[qt][t][v] * inv);
    }
  }
}

// ----------------------------- launcher ------------------------------------
extern "C" void kernel_launch(void* const* d_in, const int* in_sizes, int n_in,
                              void* d_out, int out_size, void* d_ws, size_t ws_size,
                              hipStream_t stream) {
  (void)in_sizes; (void)n_in; (void)out_size; (void)ws_size;
  const float* x    = (const float*)d_in[0];
  const float* rope = (const float*)d_in[1];
  const float* wq   = (const float*)d_in[2];
  const float* wk   = (const float*)d_in[3];
  const float* wv   = (const float*)d_in[4];
  const float* wo   = (const float*)d_in[5];
  const float* qg   = (const float*)d_in[6];
  const float* kg   = (const float*)d_in[7];
  float* out = (float*)d_out;

  const size_t LD = (size_t)L_TOK * DMODEL;   // 4,915,200
  const size_t DD = (size_t)DMODEL * DMODEL;  // 1,048,576

  char* p = (char*)d_ws;
  auto take = [&](size_t bytes) -> void* {
    void* r = (void*)p;
    p += (bytes + 255) & ~(size_t)255;
    return r;
  };
  _Float16* x_h  = (_Float16*)take(LD * 2);
  _Float16* wq_h = (_Float16*)take(DD * 2);
  _Float16* wk_h = (_Float16*)take(DD * 2);
  _Float16* wv_h = (_Float16*)take(DD * 2);
  _Float16* wo_h = (_Float16*)take(DD * 2);
  float*    Sbuf = (float*)take(LD * 4);      // f32 GEMM scratch (reused Q/K/V)
  _Float16* q_h  = (_Float16*)take(LD * 2);   // (head, L, 64)
  _Float16* k_h  = (_Float16*)take(LD * 2);   // (head, L, 64)
  _Float16* vt_h = (_Float16*)take(LD * 2);   // (head, 64, L)
  _Float16* oc_h = (_Float16*)take(LD * 2);   // (L, D) concat heads

  // 1) f32 -> f16 conversions
  f32_to_f16_kernel<<<(int)(LD / 256), 256, 0, stream>>>(x, x_h, (int)LD);
  f32_to_f16_kernel<<<(int)(DD / 256), 256, 0, stream>>>(wq, wq_h, (int)DD);
  f32_to_f16_kernel<<<(int)(DD / 256), 256, 0, stream>>>(wk, wk_h, (int)DD);
  f32_to_f16_kernel<<<(int)(DD / 256), 256, 0, stream>>>(wv, wv_h, (int)DD);
  f32_to_f16_kernel<<<(int)(DD / 256), 256, 0, stream>>>(wo, wo_h, (int)DD);

  const int gemmTiles  = (L_TOK / 64) * (DMODEL / 64);  // 1200
  const int gemmBlocks = gemmTiles / 8;                 // 8 waves per block
  const int postBlocks = (L_TOK * NHEADS) / 8;          // 8 waves per block

  // 2) Q = X Wq^T -> RMSNorm+RoPE (scale 1/sqrt(64) folded into Q)
  gemm_f16_nt<<<gemmBlocks, 256, 0, stream>>>(x_h, wq_h, Sbuf, L_TOK, DMODEL, DMODEL);
  qk_post_kernel<<<postBlocks, 256, 0, stream>>>(Sbuf, rope, qg, q_h, 0.125f);
  // 3) K
  gemm_f16_nt<<<gemmBlocks, 256, 0, stream>>>(x_h, wk_h, Sbuf, L_TOK, DMODEL, DMODEL);
  qk_post_kernel<<<postBlocks, 256, 0, stream>>>(Sbuf, rope, kg, k_h, 1.0f);
  // 4) V (transposed per head for contiguous B-fragments in attention)
  gemm_f16_nt<<<gemmBlocks, 256, 0, stream>>>(x_h, wv_h, Sbuf, L_TOK, DMODEL, DMODEL);
  v_trans_kernel<<<dim3(L_TOK / 32, DMODEL / 32), 256, 0, stream>>>(Sbuf, vt_h);

  // 5) flash attention: 150 query blocks x 16 heads, 4 waves/block
  flash_kernel<<<(L_TOK / 32) * NHEADS / 4, 128, 0, stream>>>(q_h, k_h, vt_h, oc_h);

  // 6) out = Ocat Wo^T (f32 straight to d_out)
  gemm_f16_nt<<<gemmBlocks, 256, 0, stream>>>(oc_h, wo_h, out, L_TOK, DMODEL, DMODEL);
}
